// token_representation_45629732553089
// MI455X (gfx1250) — matrix operand order repaired and checked by reading, not compiled, not run on hardware
//
#include <hip/hip_runtime.h>
#include <hip/hip_bf16.h>

// ---------------------------------------------------------------------------
// Token representation: word-emb + char BiLSTM (WMMA) + pretrain-emb -> tanh GEMM
// Target: gfx1250 (CDNA5), wave32, v_wmma_f32_16x16x32_f16
// All matrix operands are staged in WMMA fragment order so that every
// fragment load is one contiguous 32B (v16h) load per lane.
// ---------------------------------------------------------------------------

typedef __attribute__((ext_vector_type(16))) _Float16 v16h;
typedef __attribute__((ext_vector_type(8)))  float    v8f;

#define NTOK   16384
#define LSEQ   16
#define CHAR_D 64
#define HDIM   128
#define G4H    512      // 4*H
#define KLSTM  192      // CHAR_D + HDIM  (6 k-blocks of 32)
#define WORD_D 256
#define PRE_D  300
#define OUT_D  512
#define INFO_D 812
#define KOUT   832      // INFO_D padded to 26*32

// ---- WMMA 16x16x32 f16 lane/element patterns (ISA 7.12.2) -----------------
// B-matrix (32x16): lane holds col = lane&15, K-half = lane>>4; elements are
// K = e + 8*(e>=8) + 8*(lane>>4) -> fragment buffer stores [...][lane][e] contiguous.
__device__ __forceinline__ int kpatB(int e, int lane) {
    return e + ((e & 8) ? 8 : 0) + ((lane & 16) ? 8 : 0);
}
// A-matrix (16x32): inverse map (row m, k) -> fragment-layout index.
// Fragment buffer layout: ((kb*32 + hi*16 + m) * 16 + e), so that lane l
// reads its 16 elements contiguously at ((kb*32 + l) * 16).
__device__ __forceinline__ int fragIdxA(int m, int k) {
    int kb = k >> 5, kp = k & 31;
    int hi = (kp >> 3) & 1;          // which lane-half holds this k
    int kk = kp - (hi << 3);         // 0..7 or 16..23
    int e  = (kk < 8) ? kk : (kk - 8);
    return (((kb << 5) + (hi << 4) + m) << 4) + e;
}

__device__ __forceinline__ float sigmoidf_(float x) {
    return 1.0f / (1.0f + __expf(-x));
}
// Branch-free tanh: 1 - 2/(exp(2x)+1). Saturates correctly for |x| large
// (exp->0 gives -1, exp->inf gives +1); avoids libm tanhf's divergent paths.
__device__ __forceinline__ float tanh_fast(float x) {
    float e = __expf(2.0f * x);
    return 1.0f - 2.0f / (e + 1.0f);
}

// ---------------------------------------------------------------------------
// Kernel 0: weight prep into fragment-order f16 buffers.
//  WlF_{f,r}: [6][32][32][16]  value = (k<64 ? Wih[col][k] : Whh[col][k-64])
//             with k = kb*32+kpatB(e,lane), col = ct*16 + (lane&15)
//  bias_{f,r}[col] = bih[col] + bhh[col]
//  WoutF: [26][32][32][16]     value = (k<812 ? W_out[col][k] : 0)
// ---------------------------------------------------------------------------
__global__ __launch_bounds__(256) void k0_prep(
    const float* __restrict__ Wih_f, const float* __restrict__ Whh_f,
    const float* __restrict__ bih_f, const float* __restrict__ bhh_f,
    const float* __restrict__ Wih_r, const float* __restrict__ Whh_r,
    const float* __restrict__ bih_r, const float* __restrict__ bhh_r,
    const float* __restrict__ W_out,
    _Float16* __restrict__ WlF_f, _Float16* __restrict__ WlF_r,
    float* __restrict__ bias_f, float* __restrict__ bias_r,
    _Float16* __restrict__ WoutF)
{
    int i = blockIdx.x * 256 + threadIdx.x;

    if (i < 6 * 32 * 32 * 16) {                    // 98304 LSTM frag elements
        int e = i & 15, lane = (i >> 4) & 31, ct = (i >> 9) & 31, kb = i >> 14;
        int k = kb * 32 + kpatB(e, lane);
        int col = ct * 16 + (lane & 15);
        float vf = (k < CHAR_D) ? Wih_f[col * CHAR_D + k] : Whh_f[col * HDIM + (k - CHAR_D)];
        float vr = (k < CHAR_D) ? Wih_r[col * CHAR_D + k] : Whh_r[col * HDIM + (k - CHAR_D)];
        WlF_f[i] = (_Float16)vf;
        WlF_r[i] = (_Float16)vr;
    }
    if (i < G4H) {
        bias_f[i] = bih_f[i] + bhh_f[i];
        bias_r[i] = bih_r[i] + bhh_r[i];
    }
    if (i < 26 * 32 * 32 * 16) {                   // 425984 W_out frag elements
        int e = i & 15, lane = (i >> 4) & 31, ct = (i >> 9) & 31, kb = i >> 14;
        int k = kb * 32 + kpatB(e, lane);
        int col = ct * 16 + (lane & 15);
        WoutF[i] = (_Float16)((k < INFO_D) ? W_out[col * INFO_D + k] : 0.0f);
    }
}

// ---------------------------------------------------------------------------
// Kernel 1: masked char LSTM (direction = blockIdx.y), WMMA recurrence.
// Block = 16 tokens, 8 wave32. Wave w owns hidden units [16w, 16w+16).
// B-fragments (24 x v16h) are loop-invariant -> resident in registers.
// A = [ce_t | h_{t-1}] lives in LDS in fragment order (6 KB).
// All 6 A-fragments are batch-loaded so one s_wait_dscnt covers the whole
// 24-WMMA chain (LDS latency overlapped with matrix issue).
// ---------------------------------------------------------------------------
__global__ __launch_bounds__(256, 1) void k1_lstm(
    const int* __restrict__ char_ids, const int* __restrict__ char_lens,
    const float* __restrict__ char_table,
    const _Float16* __restrict__ WlF_f, const _Float16* __restrict__ WlF_r,
    const float* __restrict__ bias_f, const float* __restrict__ bias_r,
    float* __restrict__ csum)
{
    __shared__ _Float16 Afrag[6 * 32 * 16];        // 6144 B, fragment order

    const int tid  = threadIdx.x;
    const int lane = tid & 31;
    const int wid  = tid >> 5;
    const int dir  = blockIdx.y;
    const int tok0 = blockIdx.x * 16;

    const _Float16* WF   = dir ? WlF_r : WlF_f;
    const float*    bsrc = dir ? bias_r : bias_f;

    const int u0   = wid * 16;        // hidden-unit base this wave owns
    const int m16  = lane & 15;       // tile column (also A-frag row id)
    const int mrow = (lane >> 4) * 8; // C-layout row base

    // --- Hoist loop-invariant operands into registers -----------------------
    v16h bfrag[6][4];                  // 24 fragments = whole gate GEMM B
#pragma unroll
    for (int kb = 0; kb < 6; ++kb)
#pragma unroll
        for (int g = 0; g < 4; ++g) {
            const int ct = g * 8 + wid;            // col-tile = gate*8 + wave
            bfrag[kb][g] = *(const v16h*)(WF + (((kb * 32 + ct) * 32 + lane) << 4));
        }

    float bgate[4];
#pragma unroll
    for (int g = 0; g < 4; ++g) bgate[g] = bsrc[g * HDIM + u0 + m16];

    int lenv[8];
#pragma unroll
    for (int v = 0; v < 8; ++v) lenv[v] = char_lens[tok0 + v + mrow];

    // Zero the h region of A (k-blocks 2..5 -> frag indices [1024, 3072)).
    for (int i = tid; i < 2048; i += 256) Afrag[1024 + i] = (_Float16)0.0f;

    const int kLane = CHAR_D + u0 + m16;           // h column this lane writes
    const int hBase = fragIdxA(mrow, kLane);

    float cst[8], hst[8], hsum[8];
#pragma unroll
    for (int v = 0; v < 8; ++v) { cst[v] = 0.0f; hst[v] = 0.0f; hsum[v] = 0.0f; }

    for (int t = 0; t < LSEQ; ++t) {
        const int tt = dir ? (LSEQ - 1 - t) : t;

        __syncthreads();   // prior step's WMMA reads + h writeback complete
        // Stage char embeddings for timestep tt: one float4 gather / thread.
        {
            const int i0 = tid * 4;                // 256 threads x 4 = 16x64
            const int m  = i0 >> 6;
            const int d0 = i0 & 63;
            const int cid = char_ids[(tok0 + m) * LSEQ + tt];
            const float4 cv = *(const float4*)(char_table + cid * CHAR_D + d0);
            Afrag[fragIdxA(m, d0 + 0)] = (_Float16)cv.x;
            Afrag[fragIdxA(m, d0 + 1)] = (_Float16)cv.y;
            Afrag[fragIdxA(m, d0 + 2)] = (_Float16)cv.z;
            Afrag[fragIdxA(m, d0 + 3)] = (_Float16)cv.w;
        }
        __syncthreads();

        // Batch-load all A fragments first (12 ds_load_b128), then WMMA chain.
        v16h afr[6];
#pragma unroll
        for (int kb = 0; kb < 6; ++kb)
            afr[kb] = *(const v16h*)(Afrag + ((kb * 32 + lane) << 4));

        // Gate accumulators seeded with combined bias (per-column broadcast).
        v8f acc[4];
#pragma unroll
        for (int g = 0; g < 4; ++g) {
            const float b = bgate[g];
            v8f tb = {b, b, b, b, b, b, b, b};
            acc[g] = tb;
        }

#pragma unroll
        for (int kb = 0; kb < 6; ++kb)
#pragma unroll
            for (int g = 0; g < 4; ++g)
                acc[g] = __builtin_amdgcn_wmma_f32_16x16x32_f16(
                    false, afr[kb], false, bfrag[kb][g], (short)0, acc[g], false, false);

        // LSTM cell (gate order i,f,g,o); masked steps keep prior h,c.
#pragma unroll
        for (int v = 0; v < 8; ++v) {
            const bool msk = tt < lenv[v];
            float gi = sigmoidf_(acc[0][v]);
            float gf = sigmoidf_(acc[1][v]);
            float gg = tanh_fast(acc[2][v]);
            float go = sigmoidf_(acc[3][v]);
            float cn = gf * cst[v] + gi * gg;
            float hn = go * tanh_fast(cn);
            if (msk) { cst[v] = cn; hst[v] = hn; hsum[v] += hn; }
        }

        __syncthreads();   // all waves finished reading old h
#pragma unroll
        for (int v = 0; v < 8; ++v)
            Afrag[hBase + (v << 4)] = (_Float16)hst[v];
    }

    // char_sum[token][dir*128 + unit] = sum_t h_t * m_t
#pragma unroll
    for (int v = 0; v < 8; ++v)
        csum[(tok0 + v + mrow) * (2 * HDIM) + dir * HDIM + u0 + m16] = hsum[v];
}

// ---------------------------------------------------------------------------
// Kernel 2: out = tanh([we | char_sum | pe] @ W_out.T + b_out), WMMA GEMM.
// Block = 16 tokens; 8 waves x 4 col-tiles = 512 outputs; K = 832 (26 x 32).
// A (info rows) staged in LDS fragment order; B fragments straight from the
// pre-swizzled global buffer (L2-resident, one v16h load per fragment).
// ---------------------------------------------------------------------------
__global__ __launch_bounds__(256, 1) void k2_out(
    const int* __restrict__ word_ids, const int* __restrict__ pretrain_ids,
    const float* __restrict__ word_table, const float* __restrict__ pretrain_table,
    const float* __restrict__ csum, const _Float16* __restrict__ WoutF,
    const float* __restrict__ b_out, float* __restrict__ out)
{
    __shared__ _Float16 Ai[26 * 32 * 16];          // 26624 B, fragment order

    const int tid  = threadIdx.x;
    const int lane = tid & 31;
    const int wid  = tid >> 5;
    const int tok0 = blockIdx.x * 16;

    // Stage gathered info rows (f32 -> f16), scatter into fragment order.
    for (int i = tid; i < 16 * KOUT; i += 256) {
        const int m = i / KOUT, k = i % KOUT;
        const int token = tok0 + m;
        float v;
        if (k < WORD_D)       v = word_table[(long)word_ids[token] * WORD_D + k];
        else if (k < 512)     v = csum[token * (2 * HDIM) + (k - WORD_D)];
        else if (k < INFO_D)  v = pretrain_table[(long)pretrain_ids[token] * PRE_D + (k - 512)];
        else                  v = 0.0f;
        Ai[fragIdxA(m, k)] = (_Float16)v;
    }
    __syncthreads();

    const int m16  = lane & 15;
    const int mrow = (lane >> 4) * 8;
    v8f acc[4] = {};

#pragma unroll 2
    for (int kb = 0; kb < KOUT / 32; ++kb) {
        const v16h af = *(const v16h*)(Ai + ((kb * 32 + lane) << 4));
        if (kb + 1 < KOUT / 32)
            __builtin_prefetch(WoutF + (((kb + 1) * 32 + wid * 4) * 32 + lane) * 16, 0, 3);
#pragma unroll
        for (int tI = 0; tI < 4; ++tI) {
            const int ct = wid * 4 + tI;
            const v16h bf = *(const v16h*)(WoutF + (((kb * 32 + ct) * 32 + lane) << 4));
            acc[tI] = __builtin_amdgcn_wmma_f32_16x16x32_f16(
                false, af, false, bf, (short)0, acc[tI], false, false);
        }
    }

    // Epilogue: + b_out, tanh, f32 store.
#pragma unroll
    for (int tI = 0; tI < 4; ++tI) {
        const int col = wid * 64 + tI * 16 + m16;
        const float bo = b_out[col];
#pragma unroll
        for (int v = 0; v < 8; ++v)
            out[(tok0 + v + mrow) * OUT_D + col] = tanh_fast(acc[tI][v] + bo);
    }
}

// ---------------------------------------------------------------------------
extern "C" void kernel_launch(void* const* d_in, const int* in_sizes, int n_in,
                              void* d_out, int out_size, void* d_ws, size_t ws_size,
                              hipStream_t stream)
{
    const int*   word_ids       = (const int*)d_in[0];
    const int*   char_ids       = (const int*)d_in[1];
    const int*   char_lens      = (const int*)d_in[2];
    const int*   pretrain_ids   = (const int*)d_in[3];
    const float* word_table     = (const float*)d_in[4];
    const float* char_table     = (const float*)d_in[5];
    const float* pretrain_table = (const float*)d_in[6];
    const float* Wih_f = (const float*)d_in[7];
    const float* Whh_f = (const float*)d_in[8];
    const float* bih_f = (const float*)d_in[9];
    const float* bhh_f = (const float*)d_in[10];
    const float* Wih_r = (const float*)d_in[11];
    const float* Whh_r = (const float*)d_in[12];
    const float* bih_r = (const float*)d_in[13];
    const float* bhh_r = (const float*)d_in[14];
    const float* W_out = (const float*)d_in[15];
    const float* b_out = (const float*)d_in[16];
    float* out = (float*)d_out;

    // Workspace layout (all fragment buffers 32B aligned)
    char* ws = (char*)d_ws;
    float*    csum  = (float*)ws;                            // [16384][256] f32
    size_t off = (size_t)NTOK * 2 * HDIM * sizeof(float);
    _Float16* WlF_f = (_Float16*)(ws + off);  off += (size_t)6 * 32 * 32 * 16 * 2;
    _Float16* WlF_r = (_Float16*)(ws + off);  off += (size_t)6 * 32 * 32 * 16 * 2;
    float*    bias_f = (float*)(ws + off);    off += (size_t)G4H * 4;
    float*    bias_r = (float*)(ws + off);    off += (size_t)G4H * 4;
    _Float16* WoutF = (_Float16*)(ws + off);  off += (size_t)26 * 32 * 32 * 16 * 2;
    (void)ws_size; (void)in_sizes; (void)n_in; (void)out_size;

    // k0: weight prep (grid covers the largest range: 26*32*32*16)
    {
        int total = 26 * 32 * 32 * 16;
        int blocks = (total + 255) / 256;
        k0_prep<<<blocks, 256, 0, stream>>>(Wih_f, Whh_f, bih_f, bhh_f,
                                            Wih_r, Whh_r, bih_r, bhh_r, W_out,
                                            WlF_f, WlF_r, bias_f, bias_r, WoutF);
    }

    // k1: BiLSTM recurrence. 1024 token-blocks x 2 directions.
    {
        dim3 grid(NTOK / 16, 2);
        k1_lstm<<<grid, 256, 0, stream>>>(char_ids, char_lens, char_table,
                                          WlF_f, WlF_r, bias_f, bias_r, csum);
    }

    // k2: output GEMM + tanh.
    k2_out<<<NTOK / 16, 256, 0, stream>>>(word_ids, pretrain_ids,
                                          word_table, pretrain_table,
                                          csum, WoutF, b_out, out);
}
